// PointObPrEncoder_65678639891297
// MI455X (gfx1250) — matrix operand order repaired and checked by reading, not compiled, not run on hardware
//
#include <hip/hip_runtime.h>
#include <cstdint>

// ---------------------------------------------------------------------------
// Problem constants (from reference)
// ---------------------------------------------------------------------------
#define NOBS   72000
#define NB     9000
#define LMAX   8
#define DIM    256
#define HID    1024
#define HIDH   512    // FFN hidden processed in two halves (LDS budget)
#define DEPTH  8
#define NTOK   8
#define HEADS  8
#define DH     32
#define OUTD   2048
#define CL     512
#define HW_    1800   // H_*W_
#define W_     60

// ---------------------------------------------------------------------------
// WMMA types / helpers
// ---------------------------------------------------------------------------
typedef __attribute__((ext_vector_type(16))) __bf16 v16bf;
typedef __attribute__((ext_vector_type(8)))  float  v8f;

union Frag {
    uint4  q[2];
    v16bf  bf;
};

static __device__ __forceinline__ unsigned short f2bf(float f) {
    unsigned int u = __float_as_uint(f);
    unsigned int r = u + 0x7FFFu + ((u >> 16) & 1u);   // round-to-nearest-even
    return (unsigned short)(r >> 16);
}
static __device__ __forceinline__ float bf2f(unsigned short h) {
    return __uint_as_float(((unsigned int)h) << 16);
}

// exp2-based fast tanh -> v_exp_f32 + v_rcp_f32 (no libm branchy path)
static __device__ __forceinline__ float fast_tanh(float x) {
    x = fminf(10.f, fmaxf(-10.f, x));
    float e = __builtin_amdgcn_exp2f(x * 2.8853900817779268f);   // e^(2x)
    return (e - 1.f) * __builtin_amdgcn_rcpf(e + 1.f);
}
static __device__ __forceinline__ float gelu_f(float x) {
    float x3 = x * x * x;
    return 0.5f * x * (1.f + fast_tanh(0.7978845608028654f * (x + 0.044715f * x3)));
}
static __device__ __forceinline__ float fast_exp(float x) {      // e^x, x <= 0
    return __builtin_amdgcn_exp2f(x * 1.4426950408889634f);
}

// One 16x16 output tile, A from LDS (bf16 row-major, 16 rows at mbase),
// B from fragment-swizzled global weights. Accumulates into `acc`.
//
// A layout (contiguous k, matching B): lanes 0-15 hold K{0..15},
// lanes 16-31 hold K{16..31} of row (lane&15)  -> two b128 LDS loads.
// B fragments are pre-swizzled so each lane reads its 8 dwords contiguously.
static __device__ __forceinline__ v8f wmma_tile(
        v8f acc, const unsigned short* A, int lda, int mbase,
        const uint32_t* Wf, int ktiles_total, int s0, int nsteps,
        int t, int lane)
{
    const unsigned short* arow =
        A + (size_t)(mbase + (lane & 15)) * lda + (((lane >> 4) & 1) << 4);
    const uint32_t* wb = Wf + ((size_t)t * ktiles_total + s0) * 256 + (lane << 3);
    for (int s = 0; s < nsteps; ++s) {
        Frag a, b;
        a.q[0] = *(const uint4*)(arow);
        a.q[1] = *(const uint4*)(arow + 8);
        b.q[0] = *(const uint4*)(wb);
        b.q[1] = *(const uint4*)(wb + 4);
        acc = __builtin_amdgcn_wmma_f32_16x16x32_bf16(
                  false, a.bf, false, b.bf, (short)0, acc, false, false);
        arow += 32;
        wb   += 256;
    }
    return acc;
}

// ---------------------------------------------------------------------------
// Weight fragment swizzle: W[K][N] f32 row-major -> bf16 fragment order.
// B-operand layout: lanes 0-15: N=lane,  K = {2d, 2d+1}     (d = dword 0..7)
//                   lanes16-31: N=lane-16, K = {16+2d, 17+2d}
// dst dword index = ((t*ktiles + s)*32 + lane)*8 + d
// ---------------------------------------------------------------------------
__global__ void convert_frag(const float* __restrict__ W, uint32_t* __restrict__ dst,
                             int K, int Nn)
{
    int idx = blockIdx.x * 256 + threadIdx.x;
    int total = (K * Nn) >> 1;
    if (idx >= total) return;
    int d    = idx & 7;
    int lane = (idx >> 3) & 31;
    int rest = idx >> 8;
    int ktiles = K >> 5;
    int s = rest % ktiles;
    int t = rest / ktiles;
    int n = t * 16 + (lane & 15);
    int k = s * 32 + ((lane >> 4) & 1) * 16 + 2 * d;
    uint32_t lo = f2bf(W[(size_t)k * Nn + n]);
    uint32_t hi = f2bf(W[(size_t)(k + 1) * Nn + n]);
    dst[idx] = lo | (hi << 16);
}

// ---------------------------------------------------------------------------
// Bucket slot assignment (deterministic)
// ---------------------------------------------------------------------------
__global__ void zero_ints(int* a, int n) {
    int i = blockIdx.x * 256 + threadIdx.x;
    if (i < n) a[i] = 0;
}
__global__ void assign_phase1(const int* __restrict__ li, int* cnt, int* member) {
    int i = blockIdx.x * 256 + threadIdx.x;
    if (i >= NOBS) return;
    int f = li[i * 3 + 0] * HW_ + li[i * 3 + 1] * W_ + li[i * 3 + 2];
    int p = atomicAdd(&cnt[f], 1);
    if (p < LMAX) member[f * LMAX + p] = i;
}
__global__ void assign_phase2(const int* __restrict__ member,
                              const int* __restrict__ cnt, int* slot) {
    int f = blockIdx.x * 256 + threadIdx.x;
    if (f >= NB) return;
    int m = cnt[f]; if (m > LMAX) m = LMAX;
    int v[LMAX];
    for (int j = 0; j < m; ++j) v[j] = member[f * LMAX + j];
    for (int a = 1; a < m; ++a) {                 // insertion sort -> original order
        int key = v[a]; int b = a - 1;
        while (b >= 0 && v[b] > key) { v[b + 1] = v[b]; --b; }
        v[b + 1] = key;
    }
    for (int j = 0; j < m; ++j) slot[v[j]] = j;
}

// ---------------------------------------------------------------------------
// MLP embed: h = fc3(gelu(fc2(gelu(fc1(x))))), scatter rows into ctx buckets.
// 64 rows per block, 8 waves, WMMA.
// ---------------------------------------------------------------------------
__global__ __launch_bounds__(256)
void embed_kernel(const float* __restrict__ x, const int* __restrict__ li,
                  const float* __restrict__ mlp_b, const uint32_t* __restrict__ WMf,
                  const int* __restrict__ slot, unsigned short* __restrict__ ctxg)
{
    __shared__ __align__(16) unsigned short xb[64][DIM];
    __shared__ __align__(16) unsigned short yb[64][DIM];
    const int tid  = threadIdx.x;
    const int wave = tid >> 5;
    const int lane = tid & 31;
    const int row0 = blockIdx.x * 64;

    for (int i = tid; i < 64 * DIM; i += 256) {
        int r = i >> 8, c = i & 255;
        xb[r][c] = f2bf(x[(size_t)(row0 + r) * DIM + c]);
    }
    __syncthreads();

    for (int layer = 0; layer < 3; ++layer) {
        const unsigned short* A = (layer == 1) ? &yb[0][0] : &xb[0][0];
        unsigned short*       Y = (layer == 1) ? &xb[0][0] : &yb[0][0];
        const uint32_t* Wf = WMf + (size_t)layer * 32768;
        const float*    bb = mlp_b + layer * DIM;
        for (int tile = wave; tile < 64; tile += 8) {       // 4 mtiles x 16 ntiles
            int mt = tile & 3, t = tile >> 2;
            v8f acc = {0.f,0.f,0.f,0.f,0.f,0.f,0.f,0.f};
            acc = wmma_tile(acc, A, DIM, mt * 16, Wf, 8, 0, 8, t, lane);
            int col = t * 16 + (lane & 15);
            int rb  = mt * 16 + ((lane >> 4) & 1) * 8;
            float bias = bb[col];
            for (int r = 0; r < 8; ++r) {
                float v = acc[r] + bias;
                if (layer < 2) v = gelu_f(v);
                Y[(rb + r) * DIM + col] = f2bf(v);
            }
        }
        __syncthreads();
    }

    // scatter h rows (in yb) into ctx buckets
    {
        int r = tid >> 2, part = tid & 3;      // 4 threads per row, 64 cols each
        int gi = row0 + r;
        int f  = li[gi * 3 + 0] * HW_ + li[gi * 3 + 1] * W_ + li[gi * 3 + 2];
        int sl = slot[gi];
        if ((unsigned)sl < (unsigned)LMAX) {
            unsigned short* dst = ctxg + ((size_t)f * LMAX + sl) * DIM + part * 64;
            const unsigned short* src = &yb[r][part * 64];
            for (int c = 0; c < 64; c += 8)
                *(uint4*)(dst + c) = *(const uint4*)(src + c);
        }
    }
}

// ---------------------------------------------------------------------------
// LayerNorm over 256 cols for 32 rows -> bf16 output. 8 threads per row.
// ---------------------------------------------------------------------------
static __device__ __forceinline__ void layernorm256(
        const float* __restrict__ X, unsigned short* __restrict__ Y,
        const float* __restrict__ g, const float* __restrict__ b,
        float (*rA)[8], float (*rB)[8], int tid)
{
    int r = tid >> 3, sub = tid & 7;
    const float* xr = X + r * DIM;
    float s = 0.f, s2 = 0.f;
    for (int c = sub * 32; c < sub * 32 + 32; ++c) { float v = xr[c]; s += v; s2 += v * v; }
    rA[r][sub] = s; rB[r][sub] = s2;
    __syncthreads();
    float mean = 0.f, m2 = 0.f;
    for (int j = 0; j < 8; ++j) { mean += rA[r][j]; m2 += rB[r][j]; }
    mean *= (1.f / DIM);
    float var = m2 * (1.f / DIM) - mean * mean;
    float inv = rsqrtf(var + 1e-5f);
    unsigned short* yr = Y + r * DIM;
    for (int c = sub * 32; c < sub * 32 + 32; ++c)
        yr[c] = f2bf((xr[c] - mean) * inv * g[c] + b[c]);
    __syncthreads();
}

// ---------------------------------------------------------------------------
// Fused 8-layer transformer. 4 buckets (32 rows) per block, 8 waves.
// LDS ~146 KB -> two workgroups resident per 320 KB WGP.
// ---------------------------------------------------------------------------
__global__ __launch_bounds__(256)
void transformer_kernel(const float* __restrict__ bg_tok,
                        const float* __restrict__ ln1_g, const float* __restrict__ ln1_b,
                        const float* __restrict__ bq, const float* __restrict__ bk,
                        const float* __restrict__ bv, const float* __restrict__ bo,
                        const float* __restrict__ ln2_g, const float* __restrict__ ln2_b,
                        const float* __restrict__ b1, const float* __restrict__ b2,
                        const uint32_t* __restrict__ WQf, const uint32_t* __restrict__ WKf,
                        const uint32_t* __restrict__ WVf, const uint32_t* __restrict__ WOf,
                        const uint32_t* __restrict__ W1f, const uint32_t* __restrict__ W2f,
                        const unsigned short* __restrict__ ctxg,
                        const int* __restrict__ cnt,
                        unsigned short* __restrict__ trout)
{
    __shared__ __align__(16) float          trx[32][DIM];   // 32 KB residual stream (f32)
    __shared__ __align__(16) unsigned short ctx[32][DIM];   // 16 KB
    __shared__ __align__(16) unsigned short abuf[32][DIM];  // 16 KB (LN out / attn out)
    __shared__ __align__(16) unsigned short hbuf[32][HIDH]; // 32 KB FFN hidden half
    __shared__ __align__(16) unsigned short qb[32][DIM];    // 16 KB
    __shared__ __align__(16) unsigned short kb[32][DIM];    // 16 KB
    __shared__ __align__(16) unsigned short vb[32][DIM];    // 16 KB
    __shared__ float rA[32][8];
    __shared__ float rB[32][8];

    const int tid  = threadIdx.x;
    const int wave = tid >> 5;
    const int lane = tid & 31;
    const int bucket0 = blockIdx.x * 4;

    for (int i = tid; i < 32 * DIM; i += 256) {
        int r = i >> 8, c = i & 255;
        trx[r][c] = bg_tok[(r & 7) * DIM + c];
        ctx[r][c] = ctxg[((size_t)bucket0 * LMAX + r) * DIM + c];
    }
    __syncthreads();

    for (int l = 0; l < DEPTH; ++l) {
        const uint32_t* wq_l = WQf + (size_t)l * 32768;
        const uint32_t* wk_l = WKf + (size_t)l * 32768;
        const uint32_t* wv_l = WVf + (size_t)l * 32768;
        const uint32_t* wo_l = WOf + (size_t)l * 32768;
        const uint32_t* w1_l = W1f + (size_t)l * 131072;
        const uint32_t* w2_l = W2f + (size_t)l * 131072;

        // ---- LN1 -> abuf (bf16) ----
        layernorm256(&trx[0][0], &abuf[0][0], ln1_g + l * DIM, ln1_b + l * DIM, rA, rB, tid);

        // ---- q = LN(trx)@wq+bq ; k = ctx@wk+bk ; v = ctx@wv+bv ----
        for (int tile = wave; tile < 32; tile += 8) {
            int mt = tile & 1, t = tile >> 1;
            int col = t * 16 + (lane & 15);
            int rb  = mt * 16 + ((lane >> 4) & 1) * 8;
            v8f z = {0.f,0.f,0.f,0.f,0.f,0.f,0.f,0.f};
            v8f aq = wmma_tile(z, &abuf[0][0], DIM, mt * 16, wq_l, 8, 0, 8, t, lane);
            v8f ak = wmma_tile(z, &ctx[0][0],  DIM, mt * 16, wk_l, 8, 0, 8, t, lane);
            v8f av = wmma_tile(z, &ctx[0][0],  DIM, mt * 16, wv_l, 8, 0, 8, t, lane);
            float biq = bq[l * DIM + col], bik = bk[l * DIM + col], biv = bv[l * DIM + col];
            for (int r = 0; r < 8; ++r) {
                qb[rb + r][col] = f2bf(aq[r] + biq);
                kb[rb + r][col] = f2bf(ak[r] + bik);
                vb[rb + r][col] = f2bf(av[r] + biv);
            }
        }
        __syncthreads();

        // ---- attention (block-diagonal, VALU f32): thread = (bucket, head, qtok)
        {
            int b = tid >> 6, h = (tid >> 3) & 7, q = tid & 7;
            int rq = b * 8 + q;
            int cb = h * DH;
            int cntb = cnt[bucket0 + b];
            const float scale = 0.17677669529663687f;   // 1/sqrt(32)
            float sc[LMAX];
            float mx = -1e30f;
            for (int j = 0; j < LMAX; ++j) {
                int rk = b * 8 + j;
                float d = 0.f;
                for (int c = 0; c < DH; ++c)
                    d += bf2f(qb[rq][cb + c]) * bf2f(kb[rk][cb + c]);
                d = d * scale + ((j < cntb) ? 0.f : -1e9f);
                sc[j] = d; mx = fmaxf(mx, d);
            }
            float den = 0.f;
            for (int j = 0; j < LMAX; ++j) { sc[j] = fast_exp(sc[j] - mx); den += sc[j]; }
            float inv = __builtin_amdgcn_rcpf(den);
            float o[DH];
            for (int c = 0; c < DH; ++c) o[c] = 0.f;
            for (int j = 0; j < LMAX; ++j) {
                float p = sc[j] * inv;
                int rv = b * 8 + j;
                for (int c = 0; c < DH; ++c) o[c] += p * bf2f(vb[rv][cb + c]);
            }
            for (int c = 0; c < DH; ++c) abuf[rq][cb + c] = f2bf(o[c]);
        }
        __syncthreads();

        // ---- trx += attn_out @ wo + bo ----
        for (int tile = wave; tile < 32; tile += 8) {
            int mt = tile & 1, t = tile >> 1;
            v8f z = {0.f,0.f,0.f,0.f,0.f,0.f,0.f,0.f};
            v8f acc = wmma_tile(z, &abuf[0][0], DIM, mt * 16, wo_l, 8, 0, 8, t, lane);
            int col = t * 16 + (lane & 15);
            int rb  = mt * 16 + ((lane >> 4) & 1) * 8;
            float bio = bo[l * DIM + col];
            for (int r = 0; r < 8; ++r) trx[rb + r][col] += acc[r] + bio;
        }
        __syncthreads();

        // ---- LN2 -> abuf ----
        layernorm256(&trx[0][0], &abuf[0][0], ln2_g + l * DIM, ln2_b + l * DIM, rA, rB, tid);

        // ---- FFN in two hidden halves (512 cols each): keeps LDS <= 160 KB ----
        for (int h = 0; h < 2; ++h) {
            // hbuf = gelu(abuf @ w1[:, h*512 : h*512+512] + b1)
            for (int tile = wave; tile < 64; tile += 8) {    // 2 mtiles x 32 ntiles
                int mt = tile & 1, tl = tile >> 1;
                int tg = h * 32 + tl;                        // global ntile of w1
                v8f z = {0.f,0.f,0.f,0.f,0.f,0.f,0.f,0.f};
                v8f acc = wmma_tile(z, &abuf[0][0], DIM, mt * 16, w1_l, 8, 0, 8, tg, lane);
                int colg = tg * 16 + (lane & 15);
                int coll = tl * 16 + (lane & 15);
                int rb   = mt * 16 + ((lane >> 4) & 1) * 8;
                float bi = b1[l * HID + colg];
                for (int r = 0; r < 8; ++r)
                    hbuf[rb + r][coll] = f2bf(gelu_f(acc[r] + bi));
            }
            __syncthreads();

            // trx += hbuf @ w2[h*512 : h*512+512, :]  (+ b2 once, on first half)
            for (int tile = wave; tile < 32; tile += 8) {
                int mt = tile & 1, t = tile >> 1;
                v8f z = {0.f,0.f,0.f,0.f,0.f,0.f,0.f,0.f};
                v8f acc = wmma_tile(z, &hbuf[0][0], HIDH, mt * 16, w2_l, 32, h * 16, 16, t, lane);
                int col = t * 16 + (lane & 15);
                int rb  = mt * 16 + ((lane >> 4) & 1) * 8;
                float bi = (h == 0) ? b2[l * DIM + col] : 0.f;
                for (int r = 0; r < 8; ++r) trx[rb + r][col] += acc[r] + bi;
            }
            __syncthreads();
        }
    }

    // write final tokens: trout[b][tok*256+c] (row b token-major == reshape(NB,2048))
    for (int i = tid; i < 32 * DIM; i += 256) {
        int r = i >> 8, c = i & 255;
        trout[((size_t)bucket0 * LMAX + r) * DIM + c] = f2bf(trx[r][c]);
    }
}

// ---------------------------------------------------------------------------
// Projection: out[NB][512] = trout[NB][2048] @ proj_w + proj_b, masked.
// 64 rows/block; K chunked by 256 through LDS.
// ---------------------------------------------------------------------------
__global__ __launch_bounds__(256)
void proj_kernel(const unsigned short* __restrict__ trout,
                 const uint32_t* __restrict__ WPf,
                 const float* __restrict__ proj_b,
                 const int* __restrict__ cnt,
                 float* __restrict__ out)
{
    __shared__ __align__(16) unsigned short ab[64][DIM];
    const int tid  = threadIdx.x;
    const int wave = tid >> 5;
    const int lane = tid & 31;
    const int row0 = blockIdx.x * 64;

    for (int half = 0; half < 2; ++half) {
        v8f acc[8];
        for (int j = 0; j < 8; ++j) acc[j] = (v8f){0.f,0.f,0.f,0.f,0.f,0.f,0.f,0.f};
        for (int kc = 0; kc < 8; ++kc) {                     // 8 chunks x 256 = K 2048
            __syncthreads();
            for (int i = tid; i < 64 * DIM; i += 256) {
                int r = i >> 8, c = i & 255;
                int gr = row0 + r;
                ab[r][c] = (gr < NB)
                    ? trout[(size_t)gr * OUTD + kc * DIM + c] : (unsigned short)0;
            }
            __syncthreads();
            for (int j = 0; j < 8; ++j) {
                int tile = wave + (half * 8 + j) * 8;        // 4 mtiles x 32 ntiles
                int mt = tile & 3, t = tile >> 2;
                acc[j] = wmma_tile(acc[j], &ab[0][0], DIM, mt * 16,
                                   WPf, 64, kc * 8, 8, t, lane);
            }
        }
        for (int j = 0; j < 8; ++j) {
            int tile = wave + (half * 8 + j) * 8;
            int mt = tile & 3, t = tile >> 2;
            int col = t * 16 + (lane & 15);
            int rb  = row0 + mt * 16 + ((lane >> 4) & 1) * 8;
            float bias = proj_b[col];
            for (int r = 0; r < 8; ++r) {
                int row = rb + r;
                if (row < NB) {
                    float m = (cnt[row] > 0) ? 1.f : 0.f;
                    out[(size_t)row * CL + col] = m * (acc[j][r] + bias);
                }
            }
        }
        __syncthreads();
    }
}

// ---------------------------------------------------------------------------
// Host launcher
// ---------------------------------------------------------------------------
extern "C" void kernel_launch(void* const* d_in, const int* in_sizes, int n_in,
                              void* d_out, int out_size, void* d_ws, size_t ws_size,
                              hipStream_t stream)
{
    const float* x       = (const float*)d_in[0];
    const int*   li      = (const int*)  d_in[1];
    const float* mlp_w   = (const float*)d_in[2];
    const float* mlp_b   = (const float*)d_in[3];
    const float* bg_tok  = (const float*)d_in[4];
    const float* ln1_g   = (const float*)d_in[5];
    const float* ln1_b   = (const float*)d_in[6];
    const float* wq      = (const float*)d_in[7];
    const float* bq      = (const float*)d_in[8];
    const float* wk      = (const float*)d_in[9];
    const float* bk      = (const float*)d_in[10];
    const float* wv      = (const float*)d_in[11];
    const float* bv      = (const float*)d_in[12];
    const float* wo      = (const float*)d_in[13];
    const float* bo      = (const float*)d_in[14];
    const float* ln2_g   = (const float*)d_in[15];
    const float* ln2_b   = (const float*)d_in[16];
    const float* w1      = (const float*)d_in[17];
    const float* b1      = (const float*)d_in[18];
    const float* w2      = (const float*)d_in[19];
    const float* b2      = (const float*)d_in[20];
    const float* proj_w  = (const float*)d_in[21];
    const float* proj_b  = (const float*)d_in[22];

    char* ws = (char*)d_ws;
    size_t off = 0;
    auto take = [&](size_t nbytes) -> char* {
        char* p = ws + off;
        off = (off + nbytes + 255) & ~(size_t)255;
        return p;
    };
    uint32_t* WMf  = (uint32_t*)take((size_t)3 * 32768 * 4);
    uint32_t* WQf  = (uint32_t*)take((size_t)8 * 32768 * 4);
    uint32_t* WKf  = (uint32_t*)take((size_t)8 * 32768 * 4);
    uint32_t* WVf  = (uint32_t*)take((size_t)8 * 32768 * 4);
    uint32_t* WOf  = (uint32_t*)take((size_t)8 * 32768 * 4);
    uint32_t* W1f  = (uint32_t*)take((size_t)8 * 131072 * 4);
    uint32_t* W2f  = (uint32_t*)take((size_t)8 * 131072 * 4);
    uint32_t* WPf  = (uint32_t*)take((size_t)524288 * 4);
    unsigned short* ctxg  = (unsigned short*)take((size_t)NB * LMAX * DIM * 2);
    unsigned short* trout = (unsigned short*)take((size_t)NB * OUTD * 2);
    int* cnt    = (int*)take((size_t)NB * 4);
    int* member = (int*)take((size_t)NB * LMAX * 4);
    int* slot   = (int*)take((size_t)NOBS * 4);
    (void)ws_size; (void)in_sizes; (void)n_in; (void)out_size;

    // 1) zero counters + slots
    zero_ints<<<(NB + 255) / 256, 256, 0, stream>>>(cnt, NB);
    zero_ints<<<(NOBS + 255) / 256, 256, 0, stream>>>(slot, NOBS);

    // 2) weight fragment conversion (f32 -> swizzled bf16)
    for (int l = 0; l < 3; ++l)
        convert_frag<<<128, 256, 0, stream>>>(mlp_w + (size_t)l * 65536, WMf + (size_t)l * 32768, 256, 256);
    for (int l = 0; l < DEPTH; ++l) {
        convert_frag<<<128, 256, 0, stream>>>(wq + (size_t)l * 65536, WQf + (size_t)l * 32768, 256, 256);
        convert_frag<<<128, 256, 0, stream>>>(wk + (size_t)l * 65536, WKf + (size_t)l * 32768, 256, 256);
        convert_frag<<<128, 256, 0, stream>>>(wv + (size_t)l * 65536, WVf + (size_t)l * 32768, 256, 256);
        convert_frag<<<128, 256, 0, stream>>>(wo + (size_t)l * 65536, WOf + (size_t)l * 32768, 256, 256);
        convert_frag<<<512, 256, 0, stream>>>(w1 + (size_t)l * 262144, W1f + (size_t)l * 131072, 256, 1024);
        convert_frag<<<512, 256, 0, stream>>>(w2 + (size_t)l * 262144, W2f + (size_t)l * 131072, 1024, 256);
    }
    convert_frag<<<2048, 256, 0, stream>>>(proj_w, WPf, 2048, 512);

    // 3) deterministic bucket slot assignment
    assign_phase1<<<(NOBS + 255) / 256, 256, 0, stream>>>(li, cnt, member);
    assign_phase2<<<(NB + 255) / 256, 256, 0, stream>>>(member, cnt, slot);

    // 4) obs MLP + scatter into ctx
    embed_kernel<<<NOBS / 64, 256, 0, stream>>>(x, li, mlp_b, WMf, slot, ctxg);

    // 5) fused transformer (4 buckets per block, 2 blocks / WGP)
    transformer_kernel<<<NB / 4, 256, 0, stream>>>(
        bg_tok, ln1_g, ln1_b, bq, bk, bv, bo, ln2_g, ln2_b, b1, b2,
        WQf, WKf, WVf, WOf, W1f, W2f, ctxg, cnt, trout);

    // 6) output projection + occupancy mask
    proj_kernel<<<(NB + 63) / 64, 256, 0, stream>>>(trout, WPf, proj_b, cnt, (float*)d_out);
}
